// KSubspaceBaseModel_80049600463656
// MI455X (gfx1250) — matrix operand order repaired and checked by reading, not compiled, not run on hardware
//
#include <hip/hip_runtime.h>

#define R_  4
#define K_  16
#define D_  1024
#define DL_ 32
#define B_  2048

typedef __attribute__((ext_vector_type(16))) __bf16 v16bf;
typedef __attribute__((ext_vector_type(8)))  __bf16 v8bf;
typedef __attribute__((ext_vector_type(8)))  float  v8f;

// ---------------- kernel 1: per-row ||x||^2 ----------------
__global__ void xnorm_kernel(const float* __restrict__ x, float* __restrict__ xnorm) {
  const int wave = threadIdx.x >> 5;
  const int lane = threadIdx.x & 31;
  const int row  = blockIdx.x * 8 + wave;
  const float* xr = x + (size_t)row * D_;
  float acc = 0.f;
  #pragma unroll
  for (int i = 0; i < D_ / 32; ++i) {
    float v = xr[lane + 32 * i];
    acc = fmaf(v, v, acc);
  }
  #pragma unroll
  for (int m = 16; m >= 1; m >>= 1) acc += __shfl_xor(acc, m, 32);
  if (lane == 0) xnorm[row] = acc;
}

// ---------------- kernel 2: G = U^T U (32x32) per (r,k) ----------------
__global__ void gram_kernel(const float* __restrict__ Us, float* __restrict__ gram) {
  __shared__ float Uc[64 * DL_];
  const int rk = blockIdx.x;
  const float* Ub = Us + (size_t)rk * (D_ * DL_);
  const int t  = threadIdx.x;
  const int i  = t >> 3;         // 0..31
  const int j4 = (t & 7) * 4;    // 0,4,..,28
  float a0 = 0.f, a1 = 0.f, a2 = 0.f, a3 = 0.f;
  for (int c = 0; c < D_; c += 64) {
    __syncthreads();
    for (int idx = t; idx < 64 * DL_; idx += 256) Uc[idx] = Ub[(size_t)c * DL_ + idx];
    __syncthreads();
    #pragma unroll 4
    for (int l = 0; l < 64; ++l) {
      const float ui = Uc[l * DL_ + i];
      const float* up = &Uc[l * DL_ + j4];
      a0 = fmaf(ui, up[0], a0);
      a1 = fmaf(ui, up[1], a1);
      a2 = fmaf(ui, up[2], a2);
      a3 = fmaf(ui, up[3], a3);
    }
  }
  float* g = gram + (size_t)rk * (DL_ * DL_) + i * DL_ + j4;
  g[0] = a0; g[1] = a1; g[2] = a2; g[3] = a3;
}

// ---------------- kernel 3: WMMA z = xU, then loss ----------------
#define UT_PITCH 1032  // halves per U^T row: 1024 + 8 pad (keeps 16B alignment, spreads banks)

__device__ __forceinline__ void cvt4(v16bf& a, int o, float4 f) {
  a[o + 0] = (__bf16)f.x; a[o + 1] = (__bf16)f.y;
  a[o + 2] = (__bf16)f.z; a[o + 3] = (__bf16)f.w;
}

__device__ __forceinline__ v16bf ld_afrag(const float* ap) {
  // A fragment (16x32 bf16): lane half j<8 -> K = base + j ; j>=8 -> base + 16 + (j-8)
  v16bf a;
  cvt4(a, 0,  *(const float4*)(ap));
  cvt4(a, 4,  *(const float4*)(ap + 4));
  cvt4(a, 8,  *(const float4*)(ap + 16));
  cvt4(a, 12, *(const float4*)(ap + 20));
  return a;
}

__device__ __forceinline__ v16bf ld_bfrag(const __bf16* p) {
  v8bf lo = *(const v8bf*)(p);
  v8bf hi = *(const v8bf*)(p + 16);
  v16bf r;
  #pragma unroll
  for (int i = 0; i < 8; ++i) { r[i] = lo[i]; r[i + 8] = hi[i]; }
  return r;
}

__global__ void __launch_bounds__(256) zloss_kernel(
    const float* __restrict__ x, const float* __restrict__ Us,
    const float* __restrict__ gram, const float* __restrict__ xnorm,
    float* __restrict__ loss) {
  extern __shared__ char smem[];
  __bf16* Ut = (__bf16*)smem;            // 32 rows x UT_PITCH bf16  (66048 B)
  float*  Gs = (float*)(smem + 66048);   // 32x32 f32                (4096 B)
  float*  zs = (float*)(smem + 70144);   // 256 x 33 f32             (33792 B)

  const int t     = threadIdx.x;
  const int rk    = blockIdx.y;
  const int wave  = t >> 5;
  const int lane  = t & 31;
  const int laneL = lane & 15;
  const int hiK   = lane >> 4;   // lane-half selects K sub-block per ISA fragment maps

  // Stage U^T (fp32 -> bf16, transposed) and Gram into LDS.
  const float* Ub = Us + (size_t)rk * (D_ * DL_);
  for (int i = t; i < D_ * DL_; i += 256) {
    const int Dd = i >> 5, dd = i & 31;
    Ut[dd * UT_PITCH + Dd] = (__bf16)Ub[i];
  }
  for (int i = t; i < DL_ * DL_; i += 256) Gs[i] = gram[(size_t)rk * (DL_ * DL_) + i];
  __syncthreads();

  // Each wave owns a 32-row M tile: two A fragments share each pair of B fragments.
  const int rowbase = blockIdx.x * 256 + wave * 32;
  const float* xrow0 = x + (size_t)(rowbase + laneL) * D_;
  const float* xrow1 = x + (size_t)(rowbase + 16 + laneL) * D_;

  v8f acc00 = {};  // rows 0..15,  z cols 0..15
  v8f acc01 = {};  // rows 0..15,  z cols 16..31
  v8f acc10 = {};  // rows 16..31, z cols 0..15
  v8f acc11 = {};  // rows 16..31, z cols 16..31
  const __bf16* bcol0 = Ut + laneL * UT_PITCH + hiK * 8;
  const __bf16* bcol1 = Ut + (laneL + 16) * UT_PITCH + hiK * 8;

  #pragma unroll 2
  for (int kb = 0; kb < D_; kb += 32) {
    v16bf a0 = ld_afrag(xrow0 + kb + hiK * 8);
    v16bf a1 = ld_afrag(xrow1 + kb + hiK * 8);
    // B fragments (32x16 bf16): lane half h, j<8 -> K = 8h + j ; j>=8 -> K = 16 + 8h + (j-8), from U^T row
    v16bf b0 = ld_bfrag(bcol0 + kb);
    v16bf b1 = ld_bfrag(bcol1 + kb);
    acc00 = __builtin_amdgcn_wmma_f32_16x16x32_bf16(false, a0, false, b0, (short)0, acc00, false, false);
    acc01 = __builtin_amdgcn_wmma_f32_16x16x32_bf16(false, a0, false, b1, (short)0, acc01, false, false);
    acc10 = __builtin_amdgcn_wmma_f32_16x16x32_bf16(false, a1, false, b0, (short)0, acc10, false, false);
    acc11 = __builtin_amdgcn_wmma_f32_16x16x32_bf16(false, a1, false, b1, (short)0, acc11, false, false);
  }

  // Spill z tiles (C/D layout: VGPR i -> row i + 8*hiK, col laneL) to LDS.
  #pragma unroll
  for (int i = 0; i < 8; ++i) {
    const int zr0 = wave * 32 + i + 8 * hiK;
    const int zr1 = zr0 + 16;
    zs[zr0 * 33 + laneL]      = acc00[i];
    zs[zr0 * 33 + laneL + 16] = acc01[i];
    zs[zr1 * 33 + laneL]      = acc10[i];
    zs[zr1 * 33 + laneL + 16] = acc11[i];
  }
  __syncthreads();

  // loss = 0.5||x||^2 - ||z||^2 + 0.5 z^T G z   (one row per thread, all 256 active)
  {
    float zl[32];
    #pragma unroll
    for (int d = 0; d < 32; ++d) zl[d] = zs[t * 33 + d];
    float s1 = 0.f;
    #pragma unroll
    for (int d = 0; d < 32; ++d) s1 = fmaf(zl[d], zl[d], s1);
    float q = 0.f;
    for (int d2 = 0; d2 < 32; ++d2) {
      float y = 0.f;
      #pragma unroll 8
      for (int d1 = 0; d1 < 32; ++d1) y = fmaf(zl[d1], Gs[d1 * 32 + d2], y);
      q = fmaf(y, zl[d2], q);
    }
    const int n = blockIdx.x * 256 + t;
    loss[(size_t)rk * B_ + n] = 0.5f * xnorm[n] - s1 + 0.5f * q;
  }
}

// ---------------- kernel 4/5: deterministic min-over-k + mean-over-n ----------------
__global__ void minred_kernel(const float* __restrict__ loss, float* __restrict__ part) {
  __shared__ float sred[256];
  const int idx = blockIdx.x * 256 + threadIdx.x;  // 0..R*B-1
  const int r = idx >> 11;        // / B_
  const int n = idx & (B_ - 1);
  const float* lp = loss + (size_t)r * K_ * B_ + n;
  float m = lp[0];
  #pragma unroll
  for (int k = 1; k < K_; ++k) m = fminf(m, lp[(size_t)k * B_]);
  sred[threadIdx.x] = m;
  __syncthreads();
  for (int s = 128; s >= 1; s >>= 1) {
    if (threadIdx.x < s) sred[threadIdx.x] += sred[threadIdx.x + s];
    __syncthreads();
  }
  if (threadIdx.x == 0) part[blockIdx.x] = sred[0];  // grid is r-major: 8 blocks per r
}

__global__ void final_kernel(const float* __restrict__ part, float* __restrict__ out) {
  const int t = threadIdx.x;
  if (t < R_) {
    float s = 0.f;
    #pragma unroll
    for (int j = 0; j < 8; ++j) s += part[t * 8 + j];
    out[t] = s * (1.0f / (float)B_);
  }
}

extern "C" void kernel_launch(void* const* d_in, const int* in_sizes, int n_in,
                              void* d_out, int out_size, void* d_ws, size_t ws_size,
                              hipStream_t stream) {
  const float* x  = (const float*)d_in[0];   // (B, D) fp32
  const float* Us = (const float*)d_in[1];   // (R, K, D, DL) fp32
  float* out = (float*)d_out;                // (R,) fp32
  char* ws = (char*)d_ws;
  float* xnorm = (float*)(ws);               //   8,192 B
  float* gram  = (float*)(ws + 8192);        // 262,144 B
  float* loss  = (float*)(ws + 270336);      // 524,288 B
  float* part  = (float*)(ws + 794624);      //     128 B

  xnorm_kernel<<<B_ / 8, 256, 0, stream>>>(x, xnorm);
  gram_kernel<<<R_ * K_, 256, 0, stream>>>(Us, gram);
  dim3 grid(B_ / 256, R_ * K_);
  zloss_kernel<<<grid, 256, 103936, stream>>>(x, Us, gram, xnorm, loss);
  minred_kernel<<<(R_ * B_) / 256, 256, 0, stream>>>(loss, part);
  final_kernel<<<1, 32, 0, stream>>>(part, out);
  (void)in_sizes; (void)n_in; (void)out_size; (void)ws_size;
}
